// Decoder_29712583754154
// MI455X (gfx1250) — matrix operand order repaired and checked by reading, not compiled, not run on hardware
//
#include <hip/hip_runtime.h>
#include <hip/hip_bf16.h>

#define B_  32
#define S_  64
#define MV_ 30
#define V_  20000
#define E_  256
#define H_  512
#define L_  128

typedef __attribute__((ext_vector_type(16))) _Float16 v16h;
typedef __attribute__((ext_vector_type(8)))  _Float16 v8h;
typedef __attribute__((ext_vector_type(8)))  float    v8f;

union F16Frag { v16h v; v8h h[2]; };

__device__ __forceinline__ float sigmoidf_(float x) { return 1.0f / (1.0f + __expf(-x)); }

// ---------------------------------------------------------------------------
// 1) Dedup multi-hot embedding sum + tanh -> f16
// ---------------------------------------------------------------------------
__global__ void embed_kernel(const int* __restrict__ seq,        // (B,S,MV)
                             const float* __restrict__ EW,       // (V,E)
                             _Float16* __restrict__ emb16) {     // (B*S,E)
  __shared__ int sidx[MV_];
  const int bs = blockIdx.x;
  const int e  = threadIdx.x;
  if (e < MV_) sidx[e] = seq[bs * MV_ + e];
  __syncthreads();
  float acc = 0.0f;
  for (int j = 0; j < MV_; ++j) {
    const int v = sidx[j];
    if (v == 0) continue;                       // PAD_IDX row zeroed
    bool dup = false;
    for (int k = 0; k < j; ++k) dup |= (sidx[k] == v);
    if (dup) continue;                          // scatter .set(1) dedups
    acc += EW[(size_t)v * E_ + e];
  }
  emb16[(size_t)bs * E_ + e] = (_Float16)tanhf(acc);
}

// ---------------------------------------------------------------------------
// 2) f32 -> f16 weight conversion
// ---------------------------------------------------------------------------
__global__ void f32_to_f16_kernel(const float* __restrict__ in,
                                  _Float16* __restrict__ out, int n) {
  int i = blockIdx.x * blockDim.x + threadIdx.x;
  if (i < n) out[i] = (_Float16)in[i];
}

// ---------------------------------------------------------------------------
// 3) Register-blocked WMMA GEMM:
//    C(MxN,f32) = act( A(MxK,f16) @ Bt(NxK,f16)^T + bias )
//    block = 128 threads = 4 waves stacked in M; each wave: MT x NT tiles.
//    A frags reused across NT, B frags across MT -> wmma:vmem = MT*NT : 2(MT+NT)
// ---------------------------------------------------------------------------
template<int MT, int NT, bool SIGMOID>
__global__ void wmma_gemm_nt_kernel(const _Float16* __restrict__ A,
                                    const _Float16* __restrict__ Bt,
                                    const float* __restrict__ bias,
                                    float* __restrict__ C,
                                    int M, int N, int K) {
  const int lane = threadIdx.x & 31;
  const int wave = threadIdx.x >> 5;
  const int n0 = blockIdx.x * (NT * 16);
  const int m0 = (blockIdx.y * 4 + wave) * (MT * 16);
  if (m0 >= M || n0 >= N) return;                 // wave-uniform: EXEC stays full

  const int kaoff = (lane >> 4) << 3;             // A: K 0..7 / 8..15 (+16)
  const int kboff = (lane >> 4) << 3;             // B: K 0..7 / 8..15 (+16)
  const _Float16* pa0 = A  + (size_t)(m0 + (lane & 15)) * K + kaoff;
  const _Float16* pb0 = Bt + (size_t)(n0 + (lane & 15)) * K + kboff;

  v8f zero = {};
  v8f acc[MT][NT];
  #pragma unroll
  for (int mi = 0; mi < MT; ++mi)
    #pragma unroll
    for (int ni = 0; ni < NT; ++ni) acc[mi][ni] = zero;

  for (int k = 0; k < K; k += 32) {
    F16Frag a[MT], b[NT];
    #pragma unroll
    for (int mi = 0; mi < MT; ++mi) {
      const _Float16* pa = pa0 + (size_t)mi * 16 * K + k;
      __builtin_prefetch(pa + 32, 0, 1);          // global_prefetch_b8
      a[mi].h[0] = *(const v8h*)(pa);
      a[mi].h[1] = *(const v8h*)(pa + 16);
    }
    #pragma unroll
    for (int ni = 0; ni < NT; ++ni) {
      const _Float16* pb = pb0 + (size_t)ni * 16 * K + k;
      b[ni].h[0] = *(const v8h*)(pb);
      b[ni].h[1] = *(const v8h*)(pb + 16);
    }
    #pragma unroll
    for (int mi = 0; mi < MT; ++mi)
      #pragma unroll
      for (int ni = 0; ni < NT; ++ni)
        acc[mi][ni] = __builtin_amdgcn_wmma_f32_16x16x32_f16(
            false, a[mi].v, false, b[ni].v, (short)0, acc[mi][ni], false, false);
  }

  const int mb = (lane >> 4) << 3;                // D: row = m0+16mi+mb+r, col = n0+16ni+nc
  const int nc = lane & 15;
  #pragma unroll
  for (int mi = 0; mi < MT; ++mi) {
    #pragma unroll
    for (int ni = 0; ni < NT; ++ni) {
      const int col = n0 + ni * 16 + nc;
      const float bv = bias[col];
      #pragma unroll
      for (int r = 0; r < 8; ++r) {
        float v = acc[mi][ni][r] + bv;
        if (SIGMOID) v = sigmoidf_(v);
        C[(size_t)(m0 + mi * 16 + mb + r) * N + col] = v;
      }
    }
  }
}

// ---------------------------------------------------------------------------
// 4) Whole GRU in ONE workgroup (needs CDNA5's 320KB LDS/WGP):
//    1024 thr = 32 waves. h (32x512 f16, 32KB) + gh (32x1536 f32, 192KB) in LDS.
//    Per step: gh = h @ W_hh^T as a 32x1536x512 WMMA GEMM — wave w owns
//    2 M-tiles x 3 N-tiles (A frags via ds_load from LDS, B f16 from L2) —
//    then gates / h-update / m-reduction between workgroup barriers.
//    W_hh L2 traffic: 1.5MB/step * 64 = 96MB (vs 6.3GB naive); 1 launch vs 64.
// ---------------------------------------------------------------------------
#define GRU_THREADS 1024
__global__ void gru_all_kernel(const float* __restrict__ gi,        // (B*S,3H) incl b_ih
                               const _Float16* __restrict__ whh16,  // (3H,H)
                               const float* __restrict__ b_hh,      // (3H)
                               const float* __restrict__ z,         // (B,L)
                               const float* __restrict__ W_l2h,     // (H,L)
                               const float* __restrict__ b_l2h,     // (H)
                               const int* __restrict__ length,      // (B)
                               _Float16* __restrict__ outs16,       // (B*S,H) masked
                               float* __restrict__ m_out) {         // (B*S)
  extern __shared__ char smem[];
  float*    gh_sh = (float*)smem;                                   // 32*1536 f32
  _Float16* h16   = (_Float16*)(smem + (size_t)B_ * 3 * H_ * 4);    // 32*512 f16
  float*    msum  = (float*)(smem + (size_t)B_ * 3 * H_ * 4 + (size_t)B_ * H_ * 2);

  const int tid  = threadIdx.x;
  const int lane = tid & 31;
  const int wave = tid >> 5;                       // 0..31

  // ---- h0 = z @ W_l2h^T + b_l2h, straight into LDS as f16 ----
  for (int idx = tid; idx < B_ * H_; idx += GRU_THREADS) {
    const int b = idx >> 9, j = idx & (H_ - 1);
    float acc = b_l2h[j];
    #pragma unroll 8
    for (int k = 0; k < L_; ++k) acc += z[b * L_ + k] * W_l2h[(size_t)j * L_ + k];
    h16[idx] = (_Float16)acc;
  }
  __syncthreads();

  const int ntile0 = wave * 3;                     // 32 waves * 3 = 96 N-tiles
  const int kaoff  = (lane >> 4) << 3;
  const int kboff  = (lane >> 4) << 3;
  const int arow   = lane & 15;
  const int mb     = (lane >> 4) << 3;
  const int nc     = lane & 15;

  #pragma unroll 1
  for (int t = 0; t < S_; ++t) {
    // --- phase 1: gh = h @ W_hh^T (WMMA; A from LDS, B from L2) ---
    v8f zero = {};
    v8f acc[2][3];
    #pragma unroll
    for (int mi = 0; mi < 2; ++mi)
      #pragma unroll
      for (int ni = 0; ni < 3; ++ni) acc[mi][ni] = zero;

    #pragma unroll 2
    for (int kc = 0; kc < H_; kc += 32) {
      F16Frag a[2], b[3];
      #pragma unroll
      for (int mi = 0; mi < 2; ++mi) {
        const _Float16* pa = h16 + (size_t)(mi * 16 + arow) * H_ + kc + kaoff;
        a[mi].h[0] = *(const v8h*)(pa);            // ds_load_b128
        a[mi].h[1] = *(const v8h*)(pa + 16);
      }
      #pragma unroll
      for (int ni = 0; ni < 3; ++ni) {
        const _Float16* pb = whh16 + (size_t)((ntile0 + ni) * 16 + nc) * H_ + kc + kboff;
        b[ni].h[0] = *(const v8h*)(pb);
        b[ni].h[1] = *(const v8h*)(pb + 16);
      }
      #pragma unroll
      for (int mi = 0; mi < 2; ++mi)
        #pragma unroll
        for (int ni = 0; ni < 3; ++ni)
          acc[mi][ni] = __builtin_amdgcn_wmma_f32_16x16x32_f16(
              false, a[mi].v, false, b[ni].v, (short)0, acc[mi][ni], false, false);
    }
    #pragma unroll
    for (int mi = 0; mi < 2; ++mi)
      #pragma unroll
      for (int ni = 0; ni < 3; ++ni)
        #pragma unroll
        for (int r = 0; r < 8; ++r)
          gh_sh[(size_t)(mi * 16 + mb + r) * (3 * H_) + (ntile0 + ni) * 16 + nc] =
              acc[mi][ni][r];
    if (tid < B_) msum[tid] = 0.0f;
    __syncthreads();

    // --- phase 2: gates + h update + masked output + m partials ---
    for (int idx = tid; idx < B_ * H_; idx += GRU_THREADS) {
      const int b = idx >> 9, j = idx & (H_ - 1);
      const size_t gbase = ((size_t)b * S_ + t) * (3 * H_);
      const float ir  = gi[gbase + j];
      const float iz  = gi[gbase + H_ + j];
      const float inn = gi[gbase + 2 * H_ + j];
      const float hr = gh_sh[(size_t)b * (3 * H_) + j]           + b_hh[j];
      const float hz = gh_sh[(size_t)b * (3 * H_) + H_ + j]      + b_hh[H_ + j];
      const float hn = gh_sh[(size_t)b * (3 * H_) + 2 * H_ + j]  + b_hh[2 * H_ + j];
      const float hprev = (float)h16[idx];
      const float r  = sigmoidf_(ir + hr);
      const float zt = sigmoidf_(iz + hz);
      const float nn = tanhf(inn + r * hn);
      const float hnew = (1.0f - zt) * nn + zt * hprev;
      h16[idx] = (_Float16)hnew;                   // own element: no race
      const float ov = (t < length[b]) ? hnew : 0.0f;
      outs16[((size_t)b * S_ + t) * H_ + j] = (_Float16)ov;
      atomicAdd(&msum[b], fabsf(ov));              // ds_add_f32
    }
    __syncthreads();
    if (tid < B_) m_out[(size_t)tid * S_ + t] = (msum[tid] > 0.0f) ? 1.0f : 0.0f;
  }
}

// ---------------------------------------------------------------------------
// 5) Top-30 per (b,s) on probabilities (sigmoid monotone; p>0.5 <=> logit>0).
//    Row staged into LDS with CDNA5 async global->LDS copies (ASYNCcnt),
//    then 30 destructive argmax passes over LDS.
// ---------------------------------------------------------------------------
__global__ void topk_kernel(const float* __restrict__ P,     // (B*S,V)
                            float* __restrict__ Sout) {      // (B*S,MV)
  extern __shared__ char smem[];
  float* sp = (float*)smem;                               // V_ floats
  float* rv = (float*)(smem + (size_t)V_ * 4);            // 256 floats
  int*   ri = (int*)(smem + (size_t)V_ * 4 + 256 * 4);    // 256 ints
  const int bs = blockIdx.x, tid = threadIdx.x;
  const float* p = P + (size_t)bs * V_;

  // async stage: generic LDS pointer's low 32 bits are the LDS byte offset
  for (int i = tid * 4; i < V_; i += 256 * 4) {
    const unsigned ldsoff = (unsigned)(unsigned long long)(const void*)(sp + i);
    const unsigned long long ga = (unsigned long long)(const void*)(p + i);
    asm volatile("global_load_async_to_lds_b128 %0, %1, off"
                 :: "v"(ldsoff), "v"(ga) : "memory");
  }
  asm volatile("s_wait_asynccnt 0x0" ::: "memory");
  __syncthreads();
  if (tid < 3) sp[tid] = -1.0f;                           // exclude cols 0..2
  __syncthreads();

  for (int pass = 0; pass < MV_; ++pass) {
    float best = -2.0f; int bi = V_;
    for (int i = tid; i < V_; i += 256) {
      const float v = sp[i];
      if (v > best) { best = v; bi = i; }                 // strict > keeps lowest idx
    }
    rv[tid] = best; ri[tid] = bi;
    __syncthreads();
    for (int s = 128; s > 0; s >>= 1) {
      if (tid < s) {
        const float v2 = rv[tid + s]; const int i2 = ri[tid + s];
        if (v2 > rv[tid] || (v2 == rv[tid] && i2 < ri[tid])) { rv[tid] = v2; ri[tid] = i2; }
      }
      __syncthreads();
    }
    if (tid == 0) {
      const int idx = ri[0];
      Sout[(size_t)bs * MV_ + pass] = (rv[0] > 0.5f) ? (float)idx : 0.0f;
      sp[idx] = -1.0f;
    }
    __syncthreads();
  }
}

// ---------------------------------------------------------------------------
// Host orchestration
// ---------------------------------------------------------------------------
extern "C" void kernel_launch(void* const* d_in, const int* in_sizes, int n_in,
                              void* d_out, int out_size, void* d_ws, size_t ws_size,
                              hipStream_t stream) {
  (void)in_sizes; (void)n_in; (void)out_size; (void)ws_size;
  const float* z      = (const float*)d_in[0];
  const int*   seq    = (const int*)d_in[1];
  const int*   length = (const int*)d_in[2];
  const float* EW     = (const float*)d_in[3];
  const float* W_l2h  = (const float*)d_in[4];
  const float* b_l2h  = (const float*)d_in[5];
  const float* W_ih   = (const float*)d_in[6];
  const float* W_hh   = (const float*)d_in[7];
  const float* b_ih   = (const float*)d_in[8];
  const float* b_hh   = (const float*)d_in[9];
  const float* W_out  = (const float*)d_in[10];
  const float* b_out  = (const float*)d_in[11];

  char* w = (char*)d_ws;
  auto carve = [&](size_t bytes) { char* p = w; w += (bytes + 255) & ~(size_t)255; return p; };
  _Float16* emb16  = (_Float16*)carve((size_t)B_ * S_ * E_ * 2);      // 1 MB
  _Float16* wih16  = (_Float16*)carve((size_t)3 * H_ * E_ * 2);       // 0.75 MB
  _Float16* whh16  = (_Float16*)carve((size_t)3 * H_ * H_ * 2);       // 1.5 MB
  _Float16* wout16 = (_Float16*)carve((size_t)V_ * H_ * 2);           // 20.5 MB
  float*    gi     = (float*)carve((size_t)B_ * S_ * 3 * H_ * 4);     // 12 MB
  _Float16* outs16 = (_Float16*)carve((size_t)B_ * S_ * H_ * 2);      // 2 MB

  float* p_out = (float*)d_out;                          // (B*S, V)
  float* s_out = p_out + (size_t)B_ * S_ * V_;           // (B*S, MV)
  float* m_out = s_out + (size_t)B_ * S_ * MV_;          // (B*S)

  // 1) embeddings + weight conversions (independent)
  embed_kernel<<<B_ * S_, E_, 0, stream>>>(seq, EW, emb16);
  f32_to_f16_kernel<<<(3 * H_ * E_ + 255) / 256, 256, 0, stream>>>(W_ih, wih16, 3 * H_ * E_);
  f32_to_f16_kernel<<<(3 * H_ * H_ + 255) / 256, 256, 0, stream>>>(W_hh, whh16, 3 * H_ * H_);
  f32_to_f16_kernel<<<(V_ * H_ + 255) / 256, 256, 0, stream>>>(W_out, wout16, V_ * H_);

  // 2) batched input projection: gi = emb @ W_ih^T + b_ih  (2048x1536, K=256)
  wmma_gemm_nt_kernel<2, 4, false>
      <<<dim3((3 * H_) / 64, (B_ * S_) / 128), 128, 0, stream>>>(
      emb16, wih16, b_ih, gi, B_ * S_, 3 * H_, E_);

  // 3) full GRU (h0 + 64 recurrent WMMA steps) in one workgroup, 224KB LDS
  const size_t gru_lds = (size_t)B_ * 3 * H_ * 4 + (size_t)B_ * H_ * 2 + B_ * 4;
  gru_all_kernel<<<1, GRU_THREADS, gru_lds, stream>>>(
      gi, whh16, b_hh, z, W_l2h, b_l2h, length, outs16, m_out);

  // 4) big output GEMM + fused sigmoid (2048x20000, K=512) — HBM write bound
  wmma_gemm_nt_kernel<2, 5, true>
      <<<dim3(V_ / 80, (B_ * S_) / 128), 128, 0, stream>>>(
      outs16, wout16, b_out, p_out, B_ * S_, V_, H_);

  // 5) top-30 + threshold via async-staged LDS argmax
  topk_kernel<<<B_ * S_, 256, (size_t)V_ * 4 + 256 * 8, stream>>>(p_out, s_out);
}
